// GinNet_11381663335056
// MI455X (gfx1250) — matrix operand order repaired and checked by reading, not compiled, not run on hardware
//
#include <hip/hip_runtime.h>
#include <math.h>
#include <stdint.h>

// ---------------------------------------------------------------------------
// Types for CDNA5 WMMA (wave32): bf16 16x16x32, f32 accum
// ---------------------------------------------------------------------------
typedef __attribute__((ext_vector_type(16))) __bf16 v16bf;
typedef __attribute__((ext_vector_type(8)))  __bf16 v8bf;
typedef __attribute__((ext_vector_type(8)))  float  v8f;

// Problem constants
#define Bsz   16
#define Tch   100     // channels / time steps
#define H0    159
#define W0    132
#define WP    144     // W padded to 9 wmma tiles of 16 (3 supertiles of 48)
#define CIP   128     // in-channels padded to multiple of 32
#define NKT   40      // (10 kh * 128 ci) / 32
#define NMT   7       // ceil(100/16) M tiles (co / t)
#define FEAT  17424   // 132*132
#define FEATP 17440   // padded to multiple of 32 (545*32)
#define NKT_G 545
#define TPAD  112     // t rows padded to 7*16
#define GPAD  320     // 3*HID=300 padded to 20*16 (5 supertiles of 64)
#define HID   100

// ---------------------------------------------------------------------------
// Pack one conv weight tensor [100][100][10][1] f32 into WMMA A-operand order:
// per (mtile, ktile) a 512-element tile: lane l (0..31) holds v16bf, element e.
// A 16x32 order used here (contiguous k, same as B): m = l&15 ; k = e + ((l>=16)?16:0)
// K index global: k = kh*128 + ci  (ci padded to 128, zeros)
// ---------------------------------------------------------------------------
__global__ void pack_conv_w(const float* __restrict__ w, __bf16* __restrict__ Ap) {
    int tid = threadIdx.x;              // 512 = 32 lanes * 16 elems
    int l = tid >> 4, e = tid & 15;
    int mt = blockIdx.x;                // 0..6
    int kt = blockIdx.y;                // 0..39
    int m  = mt * 16 + (l & 15);
    int klocal = e + ((l >= 16) ? 16 : 0);
    int kin = kt * 32 + klocal;
    int kh = kin >> 7;                  // /128
    int ci = kin & 127;
    float v = (m < Tch && ci < Tch) ? w[((size_t)m * Tch + ci) * 10 + kh] : 0.0f;
    Ap[((size_t)(mt * NKT + kt) * 32 + l) * 16 + e] = (__bf16)v;
}

// Pack w_ih [300][17424] f32 -> bf16 [320][17440] zero padded
__global__ void pack_wih(const float* __restrict__ wih, __bf16* __restrict__ Wp) {
    size_t idx = (size_t)blockIdx.x * blockDim.x + threadIdx.x;
    const size_t total = (size_t)GPAD * FEATP;
    if (idx >= total) return;
    int k = (int)(idx % FEATP);
    int g = (int)(idx / FEATP);
    float v = (g < 3 * HID && k < FEAT) ? wih[(size_t)g * FEAT + k] : 0.0f;
    Wp[idx] = (__bf16)v;
}

// Pack x [16][100][159][132] f32 -> Xp [b][h][wp=144][cip=128] bf16, zero pads
__global__ void pack_x(const float* __restrict__ x, __bf16* __restrict__ Xp) {
    size_t idx = (size_t)blockIdx.x * blockDim.x + threadIdx.x;
    const size_t total = (size_t)Bsz * H0 * WP * CIP;
    if (idx >= total) return;
    int ci = (int)(idx & (CIP - 1));
    size_t r = idx >> 7;
    int wp = (int)(r % WP); r /= WP;
    int h  = (int)(r % H0);
    int b  = (int)(r / H0);
    float v = (wp < W0 && ci < Tch)
        ? x[(((size_t)b * Tch + ci) * H0 + h) * W0 + wp] : 0.0f;
    Xp[idx] = (__bf16)v;
}

// ---------------------------------------------------------------------------
// Conv-as-GEMM (kernel 10x1, VALID): one wave -> 16(co) x 48(w) at (b,ho)
// D[co][w] += A[co][k] * B[k][w], K = kh*128+ci; A fragment reused across the
// three 16-wide N sub-tiles -> 3 wmma per (kh,kc) K step, 120 wmma per wave.
// In/out activations: [b][h][WP][CIP] bf16. Writes bias-added bf16.
// ---------------------------------------------------------------------------
__global__ void conv_gemm(const __bf16* __restrict__ Ap, const float* __restrict__ bias,
                          const __bf16* __restrict__ Xin, __bf16* __restrict__ Xout,
                          int Hin) {
    const int lane = threadIdx.x;          // 0..31
    const int mt   = blockIdx.x;           // 0..6
    const int wt   = blockIdx.y;           // 0..2 (48-wide supertile)
    const int Hout = Hin - 9;
    const int bh   = blockIdx.z;           // b*Hout + ho
    const int b    = bh / Hout;
    const int ho   = bh - b * Hout;
    const int w0   = wt * 48;
    const int nlane = lane & 15;
    const int koff  = (lane >> 4) * 16;    // B: lanes>=16 hold K 16..31

    v8f acc0 = {0.f,0.f,0.f,0.f,0.f,0.f,0.f,0.f};
    v8f acc1 = acc0, acc2 = acc0;
    const v16bf* Atiles = (const v16bf*)Ap;

    for (int kh = 0; kh < 10; ++kh) {
        const __bf16* brow = Xin
            + (((size_t)(b * Hin + ho + kh) * WP) + (w0 + nlane)) * CIP + koff;
        if (kh < 9)   // prefetch next kh row (emits global_prefetch_b8)
            __builtin_prefetch(brow + (size_t)WP * CIP, 0, 1);
        #pragma unroll
        for (int kc = 0; kc < 4; ++kc) {
            int kt = kh * 4 + kc;
            v16bf a = Atiles[((size_t)mt * NKT + kt) * 32 + lane];
            v16bf b0 = *(const v16bf*)(brow + kc * 32);
            v16bf b1 = *(const v16bf*)(brow + 16 * CIP + kc * 32);
            v16bf b2 = *(const v16bf*)(brow + 32 * CIP + kc * 32);
            acc0 = __builtin_amdgcn_wmma_f32_16x16x32_bf16(false, a, false, b0, (short)0, acc0, false, false);
            acc1 = __builtin_amdgcn_wmma_f32_16x16x32_bf16(false, a, false, b1, (short)0, acc1, false, false);
            acc2 = __builtin_amdgcn_wmma_f32_16x16x32_bf16(false, a, false, b2, (short)0, acc2, false, false);
        }
    }

    // C/D layout: lane = N (w), 8 consecutive M (co) per lane
    const int cobase = mt * 16 + ((lane >> 4) * 8);
    float bb[8];
    #pragma unroll
    for (int r = 0; r < 8; ++r) {
        int co = cobase + r;
        bb[r] = (co < Tch) ? bias[co] : 0.0f;
    }
    v8f accs[3] = {acc0, acc1, acc2};
    #pragma unroll
    for (int sub = 0; sub < 3; ++sub) {
        const int w = w0 + sub * 16 + nlane;
        __bf16* orow = Xout + (((size_t)(b * Hout + ho) * WP) + w) * CIP + cobase;
        v8bf ov;
        #pragma unroll
        for (int r = 0; r < 8; ++r) {
            int co = cobase + r;
            float v = (co < Tch) ? (accs[sub][r] + bb[r]) : 0.0f;
            ov[r] = (__bf16)v;
        }
        *(v8bf*)orow = ov;
    }
}

// conv3: same math, but writes GRU input layout Y[b][t=co][f=ho*132+w] bf16
__global__ void conv3_gemm(const __bf16* __restrict__ Ap, const float* __restrict__ bias,
                           const __bf16* __restrict__ Xin, __bf16* __restrict__ Y,
                           int Hin) {
    const int lane = threadIdx.x;
    const int mt   = blockIdx.x;
    const int wt   = blockIdx.y;           // 0..2
    const int Hout = Hin - 9;              // 132
    const int bh   = blockIdx.z;
    const int b    = bh / Hout;
    const int ho   = bh - b * Hout;
    const int w0   = wt * 48;
    const int nlane = lane & 15;
    const int koff  = (lane >> 4) * 16;

    v8f acc0 = {0.f,0.f,0.f,0.f,0.f,0.f,0.f,0.f};
    v8f acc1 = acc0, acc2 = acc0;
    const v16bf* Atiles = (const v16bf*)Ap;

    for (int kh = 0; kh < 10; ++kh) {
        const __bf16* brow = Xin
            + (((size_t)(b * Hin + ho + kh) * WP) + (w0 + nlane)) * CIP + koff;
        if (kh < 9)
            __builtin_prefetch(brow + (size_t)WP * CIP, 0, 1);
        #pragma unroll
        for (int kc = 0; kc < 4; ++kc) {
            int kt = kh * 4 + kc;
            v16bf a = Atiles[((size_t)mt * NKT + kt) * 32 + lane];
            v16bf b0 = *(const v16bf*)(brow + kc * 32);
            v16bf b1 = *(const v16bf*)(brow + 16 * CIP + kc * 32);
            v16bf b2 = *(const v16bf*)(brow + 32 * CIP + kc * 32);
            acc0 = __builtin_amdgcn_wmma_f32_16x16x32_bf16(false, a, false, b0, (short)0, acc0, false, false);
            acc1 = __builtin_amdgcn_wmma_f32_16x16x32_bf16(false, a, false, b1, (short)0, acc1, false, false);
            acc2 = __builtin_amdgcn_wmma_f32_16x16x32_bf16(false, a, false, b2, (short)0, acc2, false, false);
        }
    }

    const int tbase = mt * 16 + ((lane >> 4) * 8);
    v8f accs[3] = {acc0, acc1, acc2};
    #pragma unroll
    for (int sub = 0; sub < 3; ++sub) {
        const int w = w0 + sub * 16 + nlane;
        if (w < W0) {
            const int f = ho * W0 + w;
            #pragma unroll
            for (int r = 0; r < 8; ++r) {
                int t = tbase + r;
                float v = (t < Tch) ? (accs[sub][r] + bias[t]) : 0.0f;
                Y[((size_t)b * TPAD + t) * FEATP + f] = (__bf16)v;
            }
        }
    }
}

// ---------------------------------------------------------------------------
// GRU input projection: xw[b][t][g] = Gi[b][t][:] . w_ih[g][:] + b_ih[g]
// A = Gi (t x FEATP), B = w_ih^T. One wave -> 16(t) x 64(g); the (expensive,
// two-chunk) A fragment is reused across four 16-wide B sub-tiles.
// ---------------------------------------------------------------------------
__global__ void gru_in_gemm(const __bf16* __restrict__ Gi, const __bf16* __restrict__ Wp,
                            const float* __restrict__ b_ih, float* __restrict__ xw) {
    const int lane = threadIdx.x;
    const int mt  = blockIdx.x;            // t tile 0..6
    const int ntb = blockIdx.y;            // g supertile 0..4 (64 wide)
    const int b   = blockIdx.z;            // 0..15
    const int nlane = lane & 15;
    const int hi = lane >> 4;

    v8f acc0 = {0.f,0.f,0.f,0.f,0.f,0.f,0.f,0.f};
    v8f acc1 = acc0, acc2 = acc0, acc3 = acc0;

    // A: lane holds row m = lane&15; K chunks at +hi*16 and +8+hi*16
    const __bf16* arow = Gi + ((size_t)b * TPAD + mt * 16 + nlane) * FEATP;
    const int aoff = hi * 16;
    // B: lane = column; 16 contiguous K at +hi*16
    const __bf16* brow0 = Wp + ((size_t)(ntb * 64 +  0 + nlane)) * FEATP + hi * 16;
    const __bf16* brow1 = Wp + ((size_t)(ntb * 64 + 16 + nlane)) * FEATP + hi * 16;
    const __bf16* brow2 = Wp + ((size_t)(ntb * 64 + 32 + nlane)) * FEATP + hi * 16;
    const __bf16* brow3 = Wp + ((size_t)(ntb * 64 + 48 + nlane)) * FEATP + hi * 16;

    for (int kt = 0; kt < NKT_G; ++kt) {
        int k0 = kt * 32;
        v8bf alo = *(const v8bf*)(arow + k0 + aoff);
        v8bf ahi = *(const v8bf*)(arow + k0 + 8 + aoff);
        v16bf a;
        #pragma unroll
        for (int i = 0; i < 8; ++i) { a[i] = alo[i]; a[8 + i] = ahi[i]; }
        v16bf b0 = *(const v16bf*)(brow0 + k0);
        v16bf b1 = *(const v16bf*)(brow1 + k0);
        v16bf b2 = *(const v16bf*)(brow2 + k0);
        v16bf b3 = *(const v16bf*)(brow3 + k0);
        acc0 = __builtin_amdgcn_wmma_f32_16x16x32_bf16(false, a, false, b0, (short)0, acc0, false, false);
        acc1 = __builtin_amdgcn_wmma_f32_16x16x32_bf16(false, a, false, b1, (short)0, acc1, false, false);
        acc2 = __builtin_amdgcn_wmma_f32_16x16x32_bf16(false, a, false, b2, (short)0, acc2, false, false);
        acc3 = __builtin_amdgcn_wmma_f32_16x16x32_bf16(false, a, false, b3, (short)0, acc3, false, false);
    }

    const int tbase = mt * 16 + hi * 8;
    v8f accs[4] = {acc0, acc1, acc2, acc3};
    #pragma unroll
    for (int sub = 0; sub < 4; ++sub) {
        const int g = ntb * 64 + sub * 16 + nlane;
        const float badd = (g < 3 * HID) ? b_ih[g] : 0.0f;
        #pragma unroll
        for (int r = 0; r < 8; ++r) {
            int t = tbase + r;
            xw[((size_t)b * TPAD + t) * GPAD + g] = accs[sub][r] + badd;
        }
    }
}

// ---------------------------------------------------------------------------
// Sequential GRU recurrence: one workgroup. w_hh (120KB) is staged into LDS
// via CDNA5 async global->LDS copies (ASYNCcnt), then reused for all 100 steps.
// Dynamic LDS layout: whh[30000] | h[16*100] | gh[16*300]  (~146KB < 320KB/WGP)
// ---------------------------------------------------------------------------
__global__ void gru_rec(const float* __restrict__ xw, const float* __restrict__ w_hh,
                        const float* __restrict__ b_hh, float* __restrict__ g_out) {
    extern __shared__ float smem[];
    float* whh = smem;                     // 30000 floats
    float* h   = whh + 3 * HID * HID;      // 16*100
    float* gh  = h + Bsz * HID;            // 16*300
    const int tid = threadIdx.x;           // 512

    // Async-stage w_hh (300*100 f32 = 7500 x 16B) into LDS
    {
        uint32_t lds0 = (uint32_t)(uintptr_t)whh;
        const char* src = (const char*)w_hh;
        for (int c = tid; c < 7500; c += 512) {
            uint32_t ldsoff = lds0 + (uint32_t)c * 16u;
            uint64_t gaddr  = (uint64_t)(uintptr_t)(src + (size_t)c * 16);
            asm volatile("global_load_async_to_lds_b128 %0, %1, off"
                         :: "v"(ldsoff), "v"(gaddr) : "memory");
        }
        asm volatile("s_wait_asynccnt 0x0" ::: "memory");
    }
    for (int i = tid; i < Bsz * HID; i += 512) h[i] = 0.0f;
    __syncthreads();

    for (int t = 0; t < Tch; ++t) {
        for (int i = tid; i < Bsz * 3 * HID; i += 512) {
            int bb = i / (3 * HID), g = i % (3 * HID);
            float s = b_hh[g];
            const float* wr = whh + g * HID;
            const float* hr = h + bb * HID;
            for (int k = 0; k < HID; ++k) s += hr[k] * wr[k];
            gh[bb * 3 * HID + g] = s;
        }
        __syncthreads();
        for (int i = tid; i < Bsz * HID; i += 512) {
            int bb = i / HID, j = i % HID;
            const float* xwr = xw + ((size_t)bb * TPAD + t) * GPAD;
            const float* ghr = gh + bb * 3 * HID;
            float r = 1.0f / (1.0f + expf(-(xwr[j] + ghr[j])));
            float z = 1.0f / (1.0f + expf(-(xwr[HID + j] + ghr[HID + j])));
            float n = tanhf(xwr[2 * HID + j] + r * ghr[2 * HID + j]);
            float hn = (1.0f - z) * n + z * h[bb * HID + j];
            h[bb * HID + j] = hn;
            g_out[((size_t)bb * Tch + t) * HID + j] = hn;
        }
        __syncthreads();
    }
}

// ---------------------------------------------------------------------------
// Heads: y1 = sigmoid(g.w_y1^T + b_y1) [16,100,14]; y2 = 10*tanh(...) [16,100,3]
// out = y1 flat ++ y2 flat
// ---------------------------------------------------------------------------
__global__ void heads_kernel(const float* __restrict__ g,
                             const float* __restrict__ wy1, const float* __restrict__ by1,
                             const float* __restrict__ wy2, const float* __restrict__ by2,
                             float* __restrict__ out) {
    int idx = blockIdx.x * blockDim.x + threadIdx.x;
    if (idx >= Bsz * Tch * 17) return;
    int c = idx % 17;
    int bt = idx / 17;
    const float* gr = g + (size_t)bt * HID;
    if (c < 14) {
        float s = by1[c];
        const float* wr = wy1 + (size_t)c * HID;
        for (int k = 0; k < HID; ++k) s += gr[k] * wr[k];
        out[(size_t)bt * 14 + c] = 1.0f / (1.0f + expf(-s));
    } else {
        int c2 = c - 14;
        float s = by2[c2];
        const float* wr = wy2 + (size_t)c2 * HID;
        for (int k = 0; k < HID; ++k) s += gr[k] * wr[k];
        out[(size_t)Bsz * Tch * 14 + (size_t)bt * 3 + c2] = 10.0f * tanhf(s);
    }
}

// ---------------------------------------------------------------------------
extern "C" void kernel_launch(void* const* d_in, const int* in_sizes, int n_in,
                              void* d_out, int out_size, void* d_ws, size_t ws_size,
                              hipStream_t stream) {
    (void)in_sizes; (void)n_in; (void)out_size; (void)ws_size;
    const float* x    = (const float*)d_in[0];
    const float* w1   = (const float*)d_in[1];
    const float* bc1  = (const float*)d_in[2];
    const float* w2   = (const float*)d_in[3];
    const float* bc2  = (const float*)d_in[4];
    const float* w3   = (const float*)d_in[5];
    const float* bc3  = (const float*)d_in[6];
    const float* w_ih = (const float*)d_in[7];
    const float* w_hh = (const float*)d_in[8];
    const float* b_ih = (const float*)d_in[9];
    const float* b_hh = (const float*)d_in[10];
    const float* w_y1 = (const float*)d_in[11];
    const float* b_y1 = (const float*)d_in[12];
    const float* w_y2 = (const float*)d_in[13];
    const float* b_y2 = (const float*)d_in[14];
    float* out = (float*)d_out;

    // Workspace carve-out (aligned to 256B)
    char* ws = (char*)d_ws;
    size_t off = 0;
    auto carve = [&](size_t bytes) {
        size_t r = off;
        off += (bytes + 255) & ~(size_t)255;
        return r;
    };
    const size_t szXp1 = (size_t)Bsz * H0  * WP * CIP * 2;        // 93.8 MB
    const size_t szXp2 = (size_t)Bsz * 150 * WP * CIP * 2;        // 88.5 MB
    const size_t szXp3 = (size_t)Bsz * 141 * WP * CIP * 2;        // 83.2 MB
    const size_t szAp  = (size_t)NMT * NKT * 512 * 2;             // 280 KB
    const size_t szWih = (size_t)GPAD * FEATP * 2;                // 11.2 MB
    const size_t szXW  = (size_t)Bsz * TPAD * GPAD * 4;           // 2.3 MB
    const size_t szG   = (size_t)Bsz * Tch * HID * 4;             // 0.64 MB

    __bf16* Xp1 = (__bf16*)(ws + carve(szXp1));
    __bf16* Xp2 = (__bf16*)(ws + carve(szXp2));
    __bf16* Xp3 = (__bf16*)(ws + carve(szXp3));
    __bf16* Ap1 = (__bf16*)(ws + carve(szAp));
    __bf16* Ap2 = (__bf16*)(ws + carve(szAp));
    __bf16* Ap3 = (__bf16*)(ws + carve(szAp));
    __bf16* Wih = (__bf16*)(ws + carve(szWih));
    float*  XW  = (float*)(ws + carve(szXW));
    float*  G   = (float*)(ws + carve(szG));
    // Y3 (GRU input, [16][112][17440] bf16, 62.5MB) aliases Xp1 (dead after conv1)
    __bf16* Y3  = Xp1;

    // --- pack ---
    pack_conv_w<<<dim3(NMT, NKT), dim3(512), 0, stream>>>(w1, Ap1);
    pack_conv_w<<<dim3(NMT, NKT), dim3(512), 0, stream>>>(w2, Ap2);
    pack_conv_w<<<dim3(NMT, NKT), dim3(512), 0, stream>>>(w3, Ap3);
    {
        size_t total = (size_t)GPAD * FEATP;
        pack_wih<<<dim3((unsigned)((total + 255) / 256)), dim3(256), 0, stream>>>(w_ih, Wih);
    }
    {
        size_t total = (size_t)Bsz * H0 * WP * CIP;
        pack_x<<<dim3((unsigned)((total + 255) / 256)), dim3(256), 0, stream>>>(x, Xp1);
    }

    // --- conv chain (WMMA GEMMs, 16x48 tiles/wave) ---
    conv_gemm<<<dim3(NMT, 3, Bsz * 150), dim3(32), 0, stream>>>(Ap1, bc1, Xp1, Xp2, H0);
    conv_gemm<<<dim3(NMT, 3, Bsz * 141), dim3(32), 0, stream>>>(Ap2, bc2, Xp2, Xp3, 150);
    conv3_gemm<<<dim3(NMT, 3, Bsz * 132), dim3(32), 0, stream>>>(Ap3, bc3, Xp3, Y3, 141);

    // --- GRU input projection (WMMA GEMM, 16x64 tiles/wave) ---
    gru_in_gemm<<<dim3(NMT, 5, Bsz), dim3(32), 0, stream>>>(Y3, Wih, b_ih, XW);

    // --- sequential recurrence (w_hh staged to LDS via async copies) ---
    {
        size_t lds_bytes = (size_t)(3 * HID * HID + Bsz * HID + Bsz * 3 * HID) * 4;
        gru_rec<<<dim3(1), dim3(512), lds_bytes, stream>>>(XW, w_hh, b_hh, G);
    }

    // --- output heads ---
    heads_kernel<<<dim3((Bsz * Tch * 17 + 255) / 256), dim3(256), 0, stream>>>(
        G, w_y1, b_y1, w_y2, b_y2, out);
}
